// RNNDec_3564822856206
// MI455X (gfx1250) — matrix operand order repaired and checked by reading, not compiled, not run on hardware
//
#include <hip/hip_runtime.h>

// ---------------- problem sizes ----------------
constexpr int Bn = 16;     // batch
constexpr int Sl = 2048;   // sequence length
constexpr int Hd = 1024;   // hidden
constexpr int G3 = 3 * Hd; // 3072 gate width
constexpr int NWAVES = 64; // scan waves: 1024 cols / 16

// LDS staging for scan: 48 rows (3 gates x 16 cols) of 1024 bf16,
// each row padded 2048 -> 2064 bytes (4-bank skew => conflict-free b128)
constexpr int ROWB    = 2064;
constexpr int SCANLDS = 48 * ROWB;   // 99072 bytes per (single-wave) block

// ---------------- WMMA vector types ----------------
typedef __attribute__((ext_vector_type(16))) __bf16 v16bf;
typedef __attribute__((ext_vector_type(8)))  float  v8f;

union UBf { v16bf v; uint4 q[2]; };

#define WMMA_BF16(Av, Bv, Cacc) \
    __builtin_amdgcn_wmma_f32_16x16x32_bf16(false, (Av), false, (Bv), (short)0, (Cacc), false, false)

// round-to-nearest-even f32 -> bf16
__device__ __forceinline__ unsigned short f32_to_bf16(float f) {
    union { float f; unsigned u; } x; x.f = f;
    unsigned r = x.u + 0x7FFFu + ((x.u >> 16) & 1u);
    return (unsigned short)(r >> 16);
}

__device__ __forceinline__ float fast_sigmoid(float x) {
    return 1.0f / (1.0f + __expf(-x));
}

// ---------------- init: zero barrier counter + h double buffer ----------------
__global__ void gru_init_kernel(unsigned int* ctr, unsigned short* hbuf) {
    int i = blockIdx.x * blockDim.x + threadIdx.x;
    if (i == 0) *ctr = 0u;
    if (i < 2 * Bn * Hd) hbuf[i] = 0;   // h0 = 0 (both parities)
}

// ---------------- vectorized f32 -> bf16 (8 elems / thread) ----------------
__global__ void cvt8_bf16_kernel(const float* __restrict__ src,
                                 unsigned short* __restrict__ dst, int n) {
    int i = (blockIdx.x * blockDim.x + threadIdx.x) * 8;
    if (i + 8 <= n) {
        float4 a = *(const float4*)(src + i);
        float4 b = *(const float4*)(src + i + 4);
        union { uint4 q; unsigned short s[8]; } o;
        o.s[0] = f32_to_bf16(a.x); o.s[1] = f32_to_bf16(a.y);
        o.s[2] = f32_to_bf16(a.z); o.s[3] = f32_to_bf16(a.w);
        o.s[4] = f32_to_bf16(b.x); o.s[5] = f32_to_bf16(b.y);
        o.s[6] = f32_to_bf16(b.z); o.s[7] = f32_to_bf16(b.w);
        *(uint4*)(dst + i) = o.q;
    }
}

// ---------------- x_gates precompute: (B*S,1024) x (1024,3072) ----------------
// one wave per 16x64 macro-tile; manual double-buffered operand pipeline so
// chunk k+1 loads overlap chunk k WMMAs (distinct live register sets).
__global__ __launch_bounds__(256, 2)
void xgate_gemm_kernel(const unsigned short* __restrict__ henc, // (B*S, Hd) bf16
                       const unsigned short* __restrict__ wih,  // (3H, Hd) bf16
                       const float* __restrict__ bih,           // (3H) f32
                       float* __restrict__ xg)                  // (S, B, 3H) f32
{
    const int wid  = blockIdx.x * (blockDim.x >> 5) + (threadIdx.x >> 5);
    const int NQT  = G3 / 64;           // 48 quad column tiles
    const int tr   = wid / NQT;         // row tile (0..2047)
    const int tq   = wid % NQT;         // quad col tile (0..47)
    const int lane = threadIdx.x & 31;
    const int cn   = lane & 15;
    const int hi   = lane >> 4;         // upper half-wave flag

    const int nb = tq * 64 + cn;        // base output column
    const int ra = tr * 16 + cn;        // A row this lane streams

    const unsigned short* arow = henc + (size_t)ra * Hd;
    const unsigned short* b0   = wih + (size_t)(nb +  0) * Hd;  // B col n == W_ih row n
    const unsigned short* b1   = wih + (size_t)(nb + 16) * Hd;
    const unsigned short* b2   = wih + (size_t)(nb + 32) * Hd;
    const unsigned short* b3   = wih + (size_t)(nb + 48) * Hd;
    const int aklo = hi ? 16 : 0, akhi = hi ? 24 : 8, bk = hi ? 16 : 0;

    v8f acc0 = {}, acc1 = {}, acc2 = {}, acc3 = {};
    UBf A0, A1, B00, B01, B02, B03, B10, B11, B12, B13;

#define LDA(Ad, k)     { Ad.q[0] = *(const uint4*)(arow + (k) + aklo);          \
                         Ad.q[1] = *(const uint4*)(arow + (k) + akhi); }
#define LDB(Bd, bp, k) { Bd.q[0] = *(const uint4*)((bp) + (k) + bk);            \
                         Bd.q[1] = *(const uint4*)((bp) + (k) + bk + 8); }

    LDA(A0, 0)
    LDB(B00, b0, 0) LDB(B01, b1, 0) LDB(B02, b2, 0) LDB(B03, b3, 0)
    #pragma unroll 4
    for (int kc = 0; kc < 16; ++kc) {          // 16 double-chunks of K=64
        const int kb = kc * 64 + 32;
        const int kn = kc * 64 + 64;           // last-iter prefetch is past-end
        LDA(A1, kb)                            // but stays inside d_ws: harmless
        LDB(B10, b0, kb) LDB(B11, b1, kb) LDB(B12, b2, kb) LDB(B13, b3, kb)
        acc0 = WMMA_BF16(A0.v, B00.v, acc0);
        acc1 = WMMA_BF16(A0.v, B01.v, acc1);
        acc2 = WMMA_BF16(A0.v, B02.v, acc2);
        acc3 = WMMA_BF16(A0.v, B03.v, acc3);
        LDA(A0, kn)
        LDB(B00, b0, kn) LDB(B01, b1, kn) LDB(B02, b2, kn) LDB(B03, b3, kn)
        acc0 = WMMA_BF16(A1.v, B10.v, acc0);
        acc1 = WMMA_BF16(A1.v, B11.v, acc1);
        acc2 = WMMA_BF16(A1.v, B12.v, acc2);
        acc3 = WMMA_BF16(A1.v, B13.v, acc3);
    }
#undef LDA
#undef LDB

    // epilogue: D layout -> xg (S, B, 3H)
    const float bias0 = bih[nb], bias1 = bih[nb + 16],
                bias2 = bih[nb + 32], bias3 = bih[nb + 48];
    #pragma unroll
    for (int i = 0; i < 8; ++i) {
        int rm = tr * 16 + i + 8 * hi;         // global row = b*S + s
        int b  = rm >> 11;                     // / 2048
        int s  = rm & (Sl - 1);
        float* orow = xg + ((size_t)s * Bn + b) * G3;
        orow[nb +  0] = acc0[i] + bias0;
        orow[nb + 16] = acc1[i] + bias1;
        orow[nb + 32] = acc2[i] + bias2;
        orow[nb + 48] = acc3[i] + bias3;
    }
}

// ---------------- persistent GRU scan ----------------
// 64 single-wave blocks; wave w owns output columns [16w, 16w+16).
// W_hh slice staged in LDS once; steady state = ds_load_b128 (B, DScnt) +
// double-buffered global h loads (A, LOADcnt) + 96 wmma per step, then a
// device-wide release/acquire barrier.
__global__ __launch_bounds__(32, 1)
void gru_scan_kernel(const float* __restrict__ xg,              // (S, B, 3H) f32
                     const unsigned short* __restrict__ whh,    // (3H, Hd) bf16
                     const float* __restrict__ bhh,             // (3H) f32
                     float* __restrict__ out,                   // (B, S, Hd) f32
                     unsigned short* __restrict__ hbuf,         // 2 x (B x Hd) bf16
                     unsigned int* __restrict__ ctr)
{
    extern __shared__ char smem[];             // 48 rows x ROWB bytes

    const int wave = blockIdx.x;               // 0..63
    const int lane = threadIdx.x & 31;
    const int cn   = lane & 15;
    const int hi   = lane >> 4;
    const int n0   = wave * 16;
    const int col  = n0 + cn;                  // output column 0..1023

    // ---- one-time W_hh slice preload: 48 rows (g*16+r) -> padded LDS rows
    for (int t = threadIdx.x; t < 48 * 128; t += 32) {   // 128 uint4 per row
        int row = t >> 7;                      // 0..47
        int c   = t & 127;
        int g   = row >> 4;                    // gate
        int r   = row & 15;                    // col within tile
        const unsigned short* srow = whh + (size_t)(g * Hd + n0 + r) * Hd;
        uint4 v = *(const uint4*)(srow + c * 8);
        *(uint4*)(smem + row * ROWB + c * 16) = v;
    }
    __syncthreads();                           // ds writes visible before reads

    // per-lane LDS base for B operand of each gate (row = g*16 + cn)
    const int bk   = hi ? 16 : 0;
    const char* lwr = smem + (0 * 16 + cn) * ROWB + bk * 2;
    const char* lwz = smem + (1 * 16 + cn) * ROWB + bk * 2;
    const char* lwn = smem + (2 * 16 + cn) * ROWB + bk * 2;

    const int am   = cn;                       // A row streamed by this lane
    const int aklo = hi ? 16 : 0, akhi = hi ? 24 : 8;

    const float bhr = bhh[0 * Hd + col];
    const float bhz = bhh[1 * Hd + col];
    const float bhn = bhh[2 * Hd + col];

    float hprev[8];                            // this wave's own h tile (f32)
    #pragma unroll
    for (int i = 0; i < 8; ++i) hprev[i] = 0.0f;

    for (int s = 0; s < Sl; ++s) {
        const int par = s & 1;
        const unsigned short* hA = hbuf + par * (Bn * Hd) + am * Hd;

        v8f accr = {}, accz = {}, accn = {};
        UBf A0, A1, R0, Z0, N0, R1, Z1, N1;

#define LDA(Ad, k)      { Ad.q[0] = *(const uint4*)(hA + (k) + aklo);           \
                          Ad.q[1] = *(const uint4*)(hA + (k) + akhi); }
#define LDS_B(Bd, lp, k){ Bd.q[0] = *(const uint4*)((lp) + (k) * 2);            \
                          Bd.q[1] = *(const uint4*)((lp) + (k) * 2 + 16); }

        LDA(A0, 0)
        LDS_B(R0, lwr, 0) LDS_B(Z0, lwz, 0) LDS_B(N0, lwn, 0)
        #pragma unroll 4
        for (int kc = 0; kc < 16; ++kc) {      // 16 double-chunks of K=64
            const int kb = kc * 64 + 32;
            const int kn = kc * 64 + 64;       // last-iter prefetch past-end:
            LDA(A1, kb)                        // other h parity / LDS pad, safe
            LDS_B(R1, lwr, kb) LDS_B(Z1, lwz, kb) LDS_B(N1, lwn, kb)
            accr = WMMA_BF16(A0.v, R0.v, accr);
            accz = WMMA_BF16(A0.v, Z0.v, accz);
            accn = WMMA_BF16(A0.v, N0.v, accn);
            LDA(A0, kn)
            LDS_B(R0, lwr, kn) LDS_B(Z0, lwz, kn) LDS_B(N0, lwn, kn)
            accr = WMMA_BF16(A1.v, R1.v, accr);
            accz = WMMA_BF16(A1.v, Z1.v, accz);
            accn = WMMA_BF16(A1.v, N1.v, accn);
        }
#undef LDA
#undef LDS_B

        // gate math on this wave's 16x16 tile (rows i+8*hi, column col)
        const float* xstep = xg + (size_t)s * Bn * G3;
        #pragma unroll
        for (int i = 0; i < 8; ++i) {
            int m = i + 8 * hi;
            const float* xrow = xstep + (size_t)m * G3;
            float xr = xrow[0 * Hd + col];
            float xz = xrow[1 * Hd + col];
            float xn = xrow[2 * Hd + col];
            float r  = fast_sigmoid(xr + accr[i] + bhr);
            float z  = fast_sigmoid(xz + accz[i] + bhz);
            float nn = tanhf(xn + r * (accn[i] + bhn));
            float hv = (1.0f - z) * nn + z * hprev[i];
            hprev[i] = hv;
            out[((size_t)m * Sl + s) * Hd + col] = hv;
            hbuf[(1 - par) * (Bn * Hd) + m * Hd + col] = f32_to_bf16(hv);
        }

        // device-wide step barrier: release h stores, acquire before next read
        if (lane == 0)
            __hip_atomic_fetch_add(ctr, 1u, __ATOMIC_RELEASE, __HIP_MEMORY_SCOPE_AGENT);
        const unsigned target = (unsigned)(NWAVES * (s + 1));
        while (__hip_atomic_load(ctr, __ATOMIC_ACQUIRE, __HIP_MEMORY_SCOPE_AGENT) < target) {
            __builtin_amdgcn_s_sleep(1);
        }
    }
}

// ---------------- host entry ----------------
extern "C" void kernel_launch(void* const* d_in, const int* in_sizes, int n_in,
                              void* d_out, int out_size, void* d_ws, size_t ws_size,
                              hipStream_t stream) {
    const float* henc = (const float*)d_in[0];   // (B,S,H)
    const float* Wih  = (const float*)d_in[1];   // (3H,H)
    const float* Whh  = (const float*)d_in[2];   // (3H,H)
    const float* bih  = (const float*)d_in[3];   // (3H)
    const float* bhh  = (const float*)d_in[4];   // (3H)
    float* out = (float*)d_out;

    // workspace layout (256B aligned):
    //   [0]         barrier counter
    //   [256]       h double buffer  2*16*1024 bf16     (65536 B)
    //   [65792]     W_ih bf16        3072*1024          (6291456 B)
    //   [6357248]   W_hh bf16        3072*1024          (6291456 B)
    //   [12648704]  h_enc bf16       16*2048*1024       (67108864 B)
    //   [79757568]  x_gates f32      (S,B,3H)           (402653184 B)
    char* ws = (char*)d_ws;
    unsigned int*   ctr    = (unsigned int*)(ws);
    unsigned short* hbuf   = (unsigned short*)(ws + 256);
    unsigned short* wih_b  = (unsigned short*)(ws + 65792);
    unsigned short* whh_b  = (unsigned short*)(ws + 6357248);
    unsigned short* henc_b = (unsigned short*)(ws + 12648704);
    float*          xg     = (float*)(ws + 79757568);

    // 1) reset barrier + h0 (graph-safe, re-run every call)
    gru_init_kernel<<<(2 * Bn * Hd + 255) / 256, 256, 0, stream>>>(ctr, hbuf);

    // 2) bf16 conversion passes (memory-bound; weights stay resident in 192MB L2)
    const int wn = G3 * Hd;                       // 3145728
    const int hn = Bn * Sl * Hd;                  // 33554432
    cvt8_bf16_kernel<<<(wn / 8 + 255) / 256, 256, 0, stream>>>(Wih, wih_b, wn);
    cvt8_bf16_kernel<<<(wn / 8 + 255) / 256, 256, 0, stream>>>(Whh, whh_b, wn);
    cvt8_bf16_kernel<<<(hn / 8 + 255) / 256, 256, 0, stream>>>(henc, henc_b, hn);

    // 3) x_gates WMMA GEMM: 2048 x 48 macro-tiles (16x64), 8 waves/block
    const int mtiles = (Bn * Sl / 16) * (G3 / 64);    // 98304
    xgate_gemm_kernel<<<mtiles / 8, 256, 0, stream>>>(henc_b, wih_b, bih, xg);

    // 4) persistent GRU scan: 64 single-wave blocks, W_hh slice in LDS
    gru_scan_kernel<<<NWAVES, 32, SCANLDS, stream>>>(xg, whh_b, bhh, out, hbuf, ctr);
}
